// CoralSTGNCDE_53017076302235
// MI455X (gfx1250) — matrix operand order repaired and checked by reading, not compiled, not run on hardware
//
#include <hip/hip_runtime.h>
#include <hip/hip_bf16.h>

// ---------------------------------------------------------------------------
// CoralSTGNCDE on MI455X (gfx1250).
//  * dominant adj@z GEMM: bf16 WMMA (v_wmma_f32_16x16x32_bf16), split-K x8
//    for 256 workgroups (short dependent WMMA chains across 60 serial GEMMs),
//    deterministic fixed-order partial reduction (no float atomics).
//  * B panel (32x512 bf16 slice shared by all 8 waves of a block) staged into
//    LDS by the Tensor Data Mover (tensor_load_to_lds + s_wait_tensorcnt),
//    with D# pad_enable inserting 16B per 1024B row -> bank-conflict-free
//    padded stride. Fallback to cooperative loads if the builtin is absent.
//  * adj converted once to bf16 (32MB) -> L2-resident for all 60 GEMMs.
// ---------------------------------------------------------------------------

#define Tn   16
#define Nn   4096
#define In   4
#define Hn   32
#define NI   (Nn * In)        // 16384 spline channels
#define NH   (Nn * Hn)        // 131072 state elements
#define KCH  512              // K-chunk per block (split-K factor 8)
#define BPAD 8                // bf16 elems of pad per LDS B row (16 bytes)

typedef __attribute__((ext_vector_type(16))) __bf16 v16bf;
typedef __attribute__((ext_vector_type(8)))  float  v8f;
typedef __attribute__((ext_vector_type(4)))  unsigned int u32x4;
typedef __attribute__((ext_vector_type(8)))  int    i32x8;
typedef __attribute__((ext_vector_type(4)))  int    i32x4;

#if defined(__has_builtin)
#if __has_builtin(__builtin_amdgcn_tensor_load_to_lds) && \
    __has_builtin(__builtin_amdgcn_s_wait_tensorcnt)
#define USE_TDM 1
#endif
#endif

__device__ __forceinline__ unsigned short f32_to_bf16_rne(float f) {
    unsigned u = __float_as_uint(f);
    u += 0x7FFFu + ((u >> 16) & 1u);      // round-to-nearest-even
    return (unsigned short)(u >> 16);
}

// ---------------------------------------------------------------------------
// 1) adj (f32, 4096x4096 row-major) -> bf16, same layout (A matrix, M x K)
// ---------------------------------------------------------------------------
__global__ void __launch_bounds__(256)
cvt_adj_kernel(const float* __restrict__ adj, unsigned short* __restrict__ out) {
    size_t i = (size_t)blockIdx.x * 256 + threadIdx.x;
    out[i] = f32_to_bf16_rne(adj[i]);
}

// ---------------------------------------------------------------------------
// 2) Natural cubic spline coefficients: fixed tridiag (1,4,1) -> Thomas.
// ---------------------------------------------------------------------------
__global__ void __launch_bounds__(256)
spline_kernel(const float* __restrict__ x, float* __restrict__ sb,
              float* __restrict__ sc, float* __restrict__ sd) {
    int ch = blockIdx.x * 256 + threadIdx.x;
    if (ch >= NI) return;
    float y[Tn];
#pragma unroll
    for (int t = 0; t < Tn; ++t) y[t] = x[t * NI + ch];

    float rhs[Tn], cp[Tn], dp[Tn], M[Tn];
#pragma unroll
    for (int t = 1; t < Tn - 1; ++t)
        rhs[t] = 6.0f * (y[t + 1] - 2.0f * y[t] + y[t - 1]);
    cp[1] = 0.25f;  dp[1] = rhs[1] * 0.25f;
#pragma unroll
    for (int i = 2; i <= Tn - 2; ++i) {
        float denom = 4.0f - cp[i - 1];
        cp[i] = 1.0f / denom;
        dp[i] = (rhs[i] - dp[i - 1]) / denom;
    }
    M[0] = 0.0f; M[Tn - 1] = 0.0f;
    M[Tn - 2] = dp[Tn - 2];
#pragma unroll
    for (int i = Tn - 3; i >= 1; --i) M[i] = dp[i] - cp[i] * M[i + 1];

#pragma unroll
    for (int i = 0; i < Tn - 1; ++i) {
        sb[i * NI + ch] = (y[i + 1] - y[i]) - (2.0f * M[i] + M[i + 1]) * (1.0f / 6.0f);
        sc[i * NI + ch] = 0.5f * M[i];
        sd[i * NI + ch] = (M[i + 1] - M[i]) * (1.0f / 6.0f);
    }
}

// ---------------------------------------------------------------------------
// 3) Encoder: z0 = x_path[0] @ W_enc + b_enc   (N x 4)(4 x 32)
// ---------------------------------------------------------------------------
__global__ void __launch_bounds__(256)
encode_kernel(const float* __restrict__ x0, const float* __restrict__ W_enc,
              const float* __restrict__ b_enc, float* __restrict__ zcur,
              float* __restrict__ zT0) {
    int idx = blockIdx.x * 256 + threadIdx.x;   // n*32 + h
    int n = idx >> 5, h = idx & 31;
    float acc = b_enc[h];
#pragma unroll
    for (int i = 0; i < In; ++i) acc += x0[n * In + i] * W_enc[i * Hn + h];
    zcur[idx] = acc;
    zT0[idx]  = acc;
}

// ---------------------------------------------------------------------------
// 4a) Stage prep: zs = z + alpha*kprev (f32) and Bt[h][n] = bf16(zs[n][h])
// ---------------------------------------------------------------------------
__global__ void __launch_bounds__(256)
prep_stage_kernel(const float* __restrict__ z, const float* __restrict__ kprev,
                  float alpha, float* __restrict__ zs,
                  unsigned short* __restrict__ Bt) {
    int idx = blockIdx.x * 256 + threadIdx.x;   // n*32 + h
    int n = idx >> 5, h = idx & 31;
    float v = z[idx] + alpha * kprev[idx];
    zs[idx] = v;
    Bt[(size_t)h * Nn + n] = f32_to_bf16_rne(v);
}

// ---------------------------------------------------------------------------
// 4b) WMMA GEMM, split-K: znp[kc](4096x32) = A[:,kc] (bf16) x B[kc,:] (bf16)
//     grid = 32 M-groups x 8 K-chunks = 256 blocks x 256 threads.
//     B slice (32 x 512 bf16) staged to LDS once per block via TDM
//     (pad_enable: +16B per 1024B row => stride 1040B, bank-conflict-free).
//     Each wave: one 16-row M tile, two 16x16 f32 accumulators (N=32),
//     16 iterations x 2 v_wmma_f32_16x16x32_bf16.
// ---------------------------------------------------------------------------
__global__ void __launch_bounds__(256)
gemm_adj_kernel(const unsigned short* __restrict__ A,
                const unsigned short* __restrict__ Bt,
                float* __restrict__ znp) {
    __shared__ __align__(16) unsigned short sB[Hn * (KCH + BPAD)];  // 33280 B

    const int kchunk = blockIdx.x & 7;
    const int mgrp   = blockIdx.x >> 3;
    const int kbase  = kchunk * KCH;

    // ---- stage B slice (rows 0..31 of Bt, cols kbase..kbase+511) into LDS ----
#ifdef USE_TDM
    if (threadIdx.x == 0) {
        const unsigned long long gva = (unsigned long long)(Bt + kbase);
        // LDS aperture: low 32 bits of generic VA == LDS byte offset.
        // ptrtoint here also escapes sB into the side-effecting TDM builtin,
        // so LLVM must treat the LDS array as written (keeps ds_loads live).
        const unsigned lds_off = (unsigned)(unsigned long long)(void*)&sB[0];
        // D# group 0: count=1, lds_addr, global addr, type=2
        u32x4 g0;
        g0[0] = 1u;                                   // count=1, user mode
        g0[1] = lds_off;                              // lds_addr (bytes)
        g0[2] = (unsigned)(gva & 0xFFFFFFFFull);      // global_addr[31:0]
        g0[3] = (unsigned)((gva >> 32) & 0x1FFFFFFull) | 0x80000000u; // [56:32] | type=2
        // D# group 1: data_size=2B, pad_enable, pad_interval=256dw, pad_amount=4dw
        i32x8 g1;
        g1[0] = (int)((1u << 16) | (1u << 20) | (7u << 22) | (3u << 25));
        g1[1] = (int)((unsigned)Nn << 16);            // tensor_dim0 = 4096 (low16)
        g1[2] = (int)((unsigned)Hn << 16);            // dim0 hi=0 | tensor_dim1=32
        g1[3] = (int)((unsigned)KCH << 16);           // dim1 hi=0 | tile_dim0=512
        g1[4] = (int)Hn;                              // tile_dim1=32, tile_dim2=0
        g1[5] = (int)Nn;                              // tensor_dim0_stride = 4096
        g1[6] = 0;                                    // stride0 hi | stride1 lo
        g1[7] = 0;                                    // stride1 hi
        i32x4 gz4 = {0, 0, 0, 0};
        i32x8 gz8 = {0, 0, 0, 0, 0, 0, 0, 0};
        // 6-arg toolchain form: (g0, g1, g2, g3, g4, cpol)
        __builtin_amdgcn_tensor_load_to_lds(g0, g1, gz4, gz4, gz8, 0);
        __builtin_amdgcn_s_wait_tensorcnt(0);
    }
    __syncthreads();
    // Compiler-level fence: LDS contents were produced by the TDM, which LLVM
    // cannot model -- forbid folding/hoisting of the B-fragment ds_loads.
    asm volatile("" ::: "memory");
#else
    for (int i = threadIdx.x; i < Hn * (KCH / 8); i += 256) {
        int r = i >> 6, c = (i & 63) << 3;            // 64 uint4-chunks per row
        *(uint4*)&sB[r * (KCH + BPAD) + c] =
            *(const uint4*)(Bt + (size_t)r * Nn + kbase + c);
    }
    __syncthreads();
#endif

    const int lane  = threadIdx.x & 31;
    const int wave  = threadIdx.x >> 5;
    const int mtile = mgrp * 8 + wave;              // 256 M tiles
    const int m0    = mtile * 16;
    const int half  = lane >> 4;                    // 0: lanes 0-15, 1: 16-31
    const int l16   = lane & 15;

    // A fragment: lane = row m0+l16, 16 contig K [half*16 .. +15]
    const unsigned short* arow = A + (size_t)(m0 + l16) * Nn + kbase + half * 16;
    // B fragments from LDS: lane = col l16 (acc0) / l16+16 (acc1), 16 contig K
    const unsigned short* srow0 = sB + (size_t)l16        * (KCH + BPAD) + half * 16;
    const unsigned short* srow1 = sB + (size_t)(l16 + 16) * (KCH + BPAD) + half * 16;

    v8f acc0 = {}; v8f acc1 = {};

#pragma unroll 2
    for (int k = 0; k < KCH; k += 32) {
        v16bf a, b0, b1;
        ((uint4*)&a)[0]  = *(const uint4*)(arow  + k);
        ((uint4*)&a)[1]  = *(const uint4*)(arow  + k + 8);
        ((uint4*)&b0)[0] = *(const uint4*)(srow0 + k);      // ds_load_b128
        ((uint4*)&b0)[1] = *(const uint4*)(srow0 + k + 8);
        ((uint4*)&b1)[0] = *(const uint4*)(srow1 + k);
        ((uint4*)&b1)[1] = *(const uint4*)(srow1 + k + 8);
        __builtin_prefetch(arow + k + 128, 0, 3);           // WGP-scope prefetch
        acc0 = __builtin_amdgcn_wmma_f32_16x16x32_bf16(
                   false, a, false, b0, (short)0, acc0, false, false);
        acc1 = __builtin_amdgcn_wmma_f32_16x16x32_bf16(
                   false, a, false, b1, (short)0, acc1, false, false);
    }

    // C layout: VGPR r -> M = m0 + half*8 + r, N = l16 (acc0) / l16+16 (acc1)
    float* crow = znp + (size_t)kchunk * NH + (size_t)(m0 + half * 8) * Hn + l16;
#pragma unroll
    for (int r = 0; r < 8; ++r) {
        crow[r * Hn]      = acc0[r];
        crow[r * Hn + 16] = acc1[r];
    }
}

// ---------------------------------------------------------------------------
// 4b') Deterministic split-K reduction: zn = sum_{kc} znp[kc]
// ---------------------------------------------------------------------------
__global__ void __launch_bounds__(256)
reduce_kernel(const float* __restrict__ znp, float* __restrict__ zn) {
    int idx = blockIdx.x * 256 + threadIdx.x;
    float a = 0.0f;
#pragma unroll
    for (int j = 0; j < 8; ++j) a += znp[(size_t)j * NH + idx];
    zn[idx] = a;
}

// ---------------------------------------------------------------------------
// 4c) Hidden: h = relu(zn@W_gcn + zs@W_time + b_gcn + b_time)
// ---------------------------------------------------------------------------
__global__ void __launch_bounds__(256)
hidden_kernel(const float* __restrict__ zn, const float* __restrict__ zs,
              const float* __restrict__ Wg, const float* __restrict__ bg,
              const float* __restrict__ Wt, const float* __restrict__ bt,
              float* __restrict__ hbuf) {
    __shared__ float sWg[Hn * Hn], sWt[Hn * Hn];
    for (int i = threadIdx.x; i < Hn * Hn; i += 256) { sWg[i] = Wg[i]; sWt[i] = Wt[i]; }
    __syncthreads();
    int idx = blockIdx.x * 256 + threadIdx.x;   // n*32 + h
    int n = idx >> 5, h = idx & 31;
    const float* zr  = zn + (size_t)n * Hn;
    const float* zsr = zs + (size_t)n * Hn;
    float acc = bg[h] + bt[h];
#pragma unroll
    for (int k = 0; k < Hn; ++k)
        acc += zr[k] * sWg[k * Hn + h] + zsr[k] * sWt[k * Hn + h];
    hbuf[idx] = fmaxf(acc, 0.0f);
}

// ---------------------------------------------------------------------------
// 4d) k-stage: k[n,h] = sum_i (h_row @ W_proj[:,h*4+i] + b_proj) * dxdt[n,i]
// ---------------------------------------------------------------------------
__global__ void __launch_bounds__(256)
kstage_kernel(const float* __restrict__ hbuf, const float* __restrict__ Wp,
              const float* __restrict__ bp, const float* __restrict__ sb,
              const float* __restrict__ sc, const float* __restrict__ sd,
              int step, float s, float* __restrict__ kout) {
    __shared__ float sWp[Hn * Hn * In];         // 32 x 128 = 16 KB
    for (int i = threadIdx.x; i < Hn * Hn * In; i += 256) sWp[i] = Wp[i];
    __syncthreads();
    int idx = blockIdx.x * 256 + threadIdx.x;   // n*32 + h
    int n = idx >> 5, h = idx & 31;
    const float* hr = hbuf + (size_t)n * Hn;
    float acc = 0.0f;
#pragma unroll
    for (int i = 0; i < In; ++i) {
        int c = h * In + i;
        float sv = bp[c];
#pragma unroll
        for (int k = 0; k < Hn; ++k) sv += hr[k] * sWp[k * (Hn * In) + c];
        int ci = step * NI + n * In + i;
        float dx = sb[ci] + 2.0f * sc[ci] * s + 3.0f * sd[ci] * s * s;
        acc += sv * dx;
    }
    kout[idx] = acc;
}

// ---------------------------------------------------------------------------
// 4e) RK4 combine: z += (k1 + 2k2 + 2k3 + k4)/6 ; write zT[t+1]
// ---------------------------------------------------------------------------
__global__ void __launch_bounds__(256)
combine_kernel(const float* __restrict__ k1, const float* __restrict__ k2,
               const float* __restrict__ k3, const float* __restrict__ k4,
               float* __restrict__ z, float* __restrict__ zT_slice) {
    int idx = blockIdx.x * 256 + threadIdx.x;
    float v = z[idx] + (k1[idx] + 2.0f * k2[idx] + 2.0f * k3[idx] + k4[idx]) * (1.0f / 6.0f);
    z[idx] = v;
    zT_slice[idx] = v;
}

// ---------------------------------------------------------------------------
// 5) Decoder: sigmoid(relu(zT@W_d1 + b_d1)@W_d2 + b_d2), OUT=1
// ---------------------------------------------------------------------------
__global__ void __launch_bounds__(256)
decoder_kernel(const float* __restrict__ zT, const float* __restrict__ Wd1,
               const float* __restrict__ bd1, const float* __restrict__ Wd2,
               const float* __restrict__ bd2, float* __restrict__ out) {
    __shared__ float sW1[Hn * 16], sb1[16], sW2[16];
    for (int i = threadIdx.x; i < Hn * 16; i += 256) sW1[i] = Wd1[i];
    if (threadIdx.x < 16) { sb1[threadIdx.x] = bd1[threadIdx.x]; sW2[threadIdx.x] = Wd2[threadIdx.x]; }
    __syncthreads();
    int idx = blockIdx.x * 256 + threadIdx.x;   // t*N + n
    const float* zr = zT + (size_t)idx * Hn;
    float acc2 = bd2[0];
#pragma unroll
    for (int j = 0; j < 16; ++j) {
        float a = sb1[j];
#pragma unroll
        for (int k = 0; k < Hn; ++k) a += zr[k] * sW1[k * 16 + j];
        acc2 += fmaxf(a, 0.0f) * sW2[j];
    }
    out[idx] = 1.0f / (1.0f + __expf(-acc2));
}

// ---------------------------------------------------------------------------
// Host-side orchestration
// ---------------------------------------------------------------------------
static inline size_t align_up(size_t x) { return (x + 255) & ~(size_t)255; }

extern "C" void kernel_launch(void* const* d_in, const int* in_sizes, int n_in,
                              void* d_out, int out_size, void* d_ws, size_t ws_size,
                              hipStream_t stream) {
    (void)in_sizes; (void)n_in; (void)out_size; (void)ws_size;
    const float* x_path = (const float*)d_in[0];
    const float* adj    = (const float*)d_in[1];
    const float* W_enc  = (const float*)d_in[2];
    const float* b_enc  = (const float*)d_in[3];
    const float* W_gcn  = (const float*)d_in[4];
    const float* b_gcn  = (const float*)d_in[5];
    const float* W_time = (const float*)d_in[6];
    const float* b_time = (const float*)d_in[7];
    const float* W_proj = (const float*)d_in[8];
    const float* b_proj = (const float*)d_in[9];
    const float* W_d1   = (const float*)d_in[10];
    const float* b_d1   = (const float*)d_in[11];
    const float* W_d2   = (const float*)d_in[12];
    const float* b_d2   = (const float*)d_in[13];
    float* out = (float*)d_out;

    // workspace carve-up
    char* ws = (char*)d_ws;
    size_t off = 0;
    unsigned short* adj_bf = (unsigned short*)(ws + off); off = align_up(off + (size_t)Nn * Nn * 2);
    float* sb   = (float*)(ws + off); off = align_up(off + (size_t)(Tn - 1) * NI * 4);
    float* sc   = (float*)(ws + off); off = align_up(off + (size_t)(Tn - 1) * NI * 4);
    float* sd   = (float*)(ws + off); off = align_up(off + (size_t)(Tn - 1) * NI * 4);
    float* zcur = (float*)(ws + off); off = align_up(off + (size_t)NH * 4);
    float* zs   = (float*)(ws + off); off = align_up(off + (size_t)NH * 4);
    unsigned short* Bt = (unsigned short*)(ws + off); off = align_up(off + (size_t)NH * 2);
    float* znp  = (float*)(ws + off); off = align_up(off + (size_t)8 * NH * 4);   // split-K partials
    float* zn   = (float*)(ws + off); off = align_up(off + (size_t)NH * 4);
    float* hbuf = (float*)(ws + off); off = align_up(off + (size_t)NH * 4);
    float* kbuf[4];
    for (int i = 0; i < 4; ++i) { kbuf[i] = (float*)(ws + off); off = align_up(off + (size_t)NH * 4); }
    float* zT   = (float*)(ws + off); off = align_up(off + (size_t)Tn * NH * 4);

    // 1) adj -> bf16 (L2-resident for all 60 GEMMs)
    cvt_adj_kernel<<<(Nn * (size_t)Nn) / 256, 256, 0, stream>>>(adj, adj_bf);
    // 2) spline coefficients
    spline_kernel<<<NI / 256, 256, 0, stream>>>(x_path, sb, sc, sd);
    // 3) encoder -> z0 (also zT[0])
    encode_kernel<<<NH / 256, 256, 0, stream>>>(x_path, W_enc, b_enc, zcur, zT);

    const float alpha[4] = {0.0f, 0.5f, 0.5f, 1.0f};
    const float sval[4]  = {0.0f, 0.5f, 0.5f, 1.0f};

    for (int t = 0; t < Tn - 1; ++t) {
        for (int s = 0; s < 4; ++s) {
            const float* kprev = (s == 0) ? zcur : kbuf[s - 1];  // alpha=0 for s==0
            prep_stage_kernel<<<NH / 256, 256, 0, stream>>>(zcur, kprev, alpha[s], zs, Bt);
            gemm_adj_kernel<<<256, 256, 0, stream>>>(adj_bf, Bt, znp);
            reduce_kernel<<<NH / 256, 256, 0, stream>>>(znp, zn);
            hidden_kernel<<<NH / 256, 256, 0, stream>>>(zn, zs, W_gcn, b_gcn, W_time, b_time, hbuf);
            kstage_kernel<<<NH / 256, 256, 0, stream>>>(hbuf, W_proj, b_proj, sb, sc, sd,
                                                        t, sval[s], kbuf[s]);
        }
        combine_kernel<<<NH / 256, 256, 0, stream>>>(kbuf[0], kbuf[1], kbuf[2], kbuf[3],
                                                     zcur, zT + (size_t)(t + 1) * NH);
    }

    // 5) decoder over all (t, n)
    decoder_kernel<<<(Tn * Nn) / 256, 256, 0, stream>>>(zT, W_d1, b_d1, W_d2, b_d2, out);
}
